// GRU_unit_for_vae_72181220377171
// MI455X (gfx1250) — matrix operand order repaired and checked by reading, not compiled, not run on hardware
//
#include <hip/hip_runtime.h>
#include <hip/hip_bf16.h>
#include <stdint.h>

typedef __attribute__((ext_vector_type(16))) __bf16 v16bf;
typedef __attribute__((ext_vector_type(8)))  __bf16 v8bf;
typedef __attribute__((ext_vector_type(4)))  __bf16 v4bf;
typedef __attribute__((ext_vector_type(8)))  float  v8f;
typedef __attribute__((ext_vector_type(4)))  float  v4f;
typedef __attribute__((ext_vector_type(4)))  unsigned int v4u;
typedef __attribute__((ext_vector_type(4)))  int v4i;
typedef __attribute__((ext_vector_type(8)))  int v8i;

namespace {
constexpr int kLatent       = 128;
constexpr int kInput        = 256;
constexpr int kLI           = 384;   // latent + input
constexpr int kUnits        = 128;
constexpr int kRowsPerBlock = 128;   // 8 waves x 16 rows
constexpr int kThreads      = 256;

// LDS strides (bf16 elements), padded so 16 lanes x 16B land on disjoint bank groups
constexpr int kYS  = 392;  // y buffer row stride      (392/2 = 196 banks == 4 mod 64)
constexpr int kWS  = 392;  // layer-1 weight K stride
constexpr int kW2S = 136;  // layer-2 weight K stride  (68 banks == 4 mod 64)
constexpr int kHS  = 136;  // layer-1 activation stride

constexpr unsigned kW1Bytes = (unsigned)kUnits * kWS * 2u;   // 100352 B per layer-1 matrix
constexpr unsigned kW2Bytes = (unsigned)kUnits * kW2S * 2u;  //  34816 B per layer-2 matrix

// Workspace layout (bf16 elements): pre-converted, transposed [N][K] weight images
constexpr int oWu1 = 0;
constexpr int oWr1 = oWu1 + kUnits * kWS;
constexpr int oWn1 = oWr1 + kUnits * kWS;
constexpr int oWu2 = oWn1 + kUnits * kWS;
constexpr int oWr2 = oWu2 + kUnits * kW2S;
constexpr int oWn2 = oWr2 + kUnits * kW2S;

// LDS: yBuf(100352) + wBuf(100352) + w2Buf(34816) + h1Buf(34816) = 270336 B of 320 KB
constexpr size_t kShmemBytes =
    (size_t)kRowsPerBlock * kYS * 2 + (size_t)kUnits * kWS * 2 +
    (size_t)kUnits * kW2S * 2 + (size_t)kRowsPerBlock * kHS * 2;
}

#if __has_builtin(__builtin_amdgcn_tensor_load_to_lds) && __has_builtin(__builtin_amdgcn_s_wait_tensorcnt)
#define GRU_USE_TDM 1
#else
#define GRU_USE_TDM 0
#endif

static __device__ __forceinline__ float sigmoidf_(float v) {
  return 1.0f / (1.0f + __expf(-v));
}

static __device__ __forceinline__ float fast_tanhf_(float v) {
#if __has_builtin(__builtin_amdgcn_tanhf)
  return __builtin_amdgcn_tanhf(v);    // gfx1250 v_tanh_f32 (TRANS, co-executes with WMMA)
#else
  return tanhf(v);
#endif
}

#if GRU_USE_TDM
// Issue a flat 1-D TDM copy: global (pre-formatted bf16 image) -> LDS.
// D# per CDNA5 ISA: group0 = {count/flags, lds_addr, global_addr, type=2},
// group1 = {data_size=8B, tensor_dim0=units, tensor_dim1=1, tile_dim0=units, tile_dim1=1, stride}.
static __device__ __forceinline__ void tdm_load(const void* gsrc, void* ldsDst, unsigned bytes) {
  const unsigned long long ga = (unsigned long long)(uintptr_t)gsrc;
  const unsigned lds   = (unsigned)(uintptr_t)ldsDst;   // addr[31:0] == LDS byte address
  const unsigned units = bytes >> 3;                    // 8-byte elements (<= 65535)
  v4u g0;
  g0[0] = 1u;                                                   // count = 1 (valid D#)
  g0[1] = lds;                                                  // lds_addr
  g0[2] = (unsigned)ga;                                         // global_addr[31:0]
  g0[3] = (unsigned)((ga >> 32) & 0x01FFFFFFu) | 0x80000000u;   // addr[56:32] | type=2
  v8i g1;
  g1[0] = (int)(3u << 16);                        // data_size = 3 (8 bytes)
  g1[1] = (int)((units & 0xFFFFu) << 16);         // tensor_dim0[15:0]
  g1[2] = (int)((units >> 16) | (1u << 16));      // tensor_dim0[31:16] | tensor_dim1 = 1
  g1[3] = (int)((units & 0xFFFFu) << 16);         // tile_dim0 = units
  g1[4] = 1;                                      // tile_dim1 = 1
  g1[5] = (int)units;                             // tensor_dim0_stride[31:0]
  g1[6] = 0;
  g1[7] = 0;
  const v4i z4 = (v4i){0, 0, 0, 0};
#if __clang_major__ >= 23
  const v8i z8 = (v8i){0, 0, 0, 0, 0, 0, 0, 0};
  __builtin_amdgcn_tensor_load_to_lds(g0, g1, z4, z4, z8, 0);
#else
  __builtin_amdgcn_tensor_load_to_lds(g0, g1, z4, z4, 0);
#endif
}
#endif

// Stage one weight image into LDS. TDM path: wave 0 issues the DMA (TENSORcnt-tracked).
// Fallback: cooperative 16B vector copy (same barrier placement works for both).
static __device__ __forceinline__ void stage_weight(__bf16* ldsDst, const __bf16* gsrc,
                                                    unsigned bytes, int tid) {
#if GRU_USE_TDM
  if (tid < 32) tdm_load(gsrc, ldsDst, bytes);
#else
  v4u* d = (v4u*)ldsDst;
  const v4u* s = (const v4u*)gsrc;
  for (unsigned i = (unsigned)tid; i < (bytes >> 4); i += kThreads) d[i] = s[i];
#endif
}

static __device__ __forceinline__ void weight_fence(int tid) {
#if GRU_USE_TDM
  if (tid < 32) __builtin_amdgcn_s_wait_tensorcnt(0);
#endif
  (void)tid;
}

static __device__ __forceinline__ v16bf concat8(v8bf lo, v8bf hi) {
  return __builtin_shufflevector(lo, hi, 0,1,2,3,4,5,6,7,8,9,10,11,12,13,14,15);
}

// A fragment (16x32 bf16), ISA 7.12.2 layout: lanes 0-15 row M=lane, K chunks [k0..k0+7],
// [k0+16..k0+23]; lanes 16-31 same rows, chunks shifted by +8.
static __device__ __forceinline__ v16bf load_a_frag(const __bf16* __restrict__ buf,
                                                    int stride, int row0, int k0, int lane) {
  const int r  = row0 + (lane & 15);
  const int kb = k0 + ((lane >> 4) << 3);
  const __bf16* p = buf + r * stride + kb;
  return concat8(*(const v8bf*)p, *(const v8bf*)(p + 16));      // 2x ds_load_b128
}

// B fragment (32x16 bf16) from K-contiguous transposed weights: lane holds col N = lane&15,
// K chunks [kb..kb+7], [kb+16..kb+23] with kb = k0 + 8*(lane>=16) (matches A layout).
static __device__ __forceinline__ v16bf load_b_frag(const __bf16* __restrict__ wT,
                                                    int stride, int n0, int k0, int lane) {
  const int n  = n0 + (lane & 15);
  const int kb = k0 + ((lane >> 4) << 3);
  const __bf16* p = wT + n * stride + kb;
  return concat8(*(const v8bf*)p, *(const v8bf*)(p + 16));      // 2x ds_load_b128
}

// One wave: [16 x 32*KTILES] @ [32*KTILES x 128] + bias -> acc[8] (8 N-tiles of 16)
template<int KTILES>
static __device__ __forceinline__ void gemm_16xN(const __bf16* __restrict__ aBuf, int as,
                                                 const __bf16* __restrict__ wT, int ws,
                                                 const float* __restrict__ bias,
                                                 int rowBase, int lane, v8f acc[8]) {
#pragma unroll
  for (int nt = 0; nt < 8; ++nt) {
    const float b = bias[nt * 16 + (lane & 15)];
#pragma unroll
    for (int j = 0; j < 8; ++j) acc[nt][j] = b;
  }
  for (int kt = 0; kt < KTILES; ++kt) {
    const v16bf a = load_a_frag(aBuf, as, rowBase, kt * 32, lane);
#pragma unroll
    for (int nt = 0; nt < 8; ++nt) {
      const v16bf b = load_b_frag(wT, ws, nt * 16, kt * 32, lane);
      acc[nt] = __builtin_amdgcn_wmma_f32_16x16x32_bf16(
          false, a, false, b, (short)0, acc[nt], false, false);
    }
  }
}

// ---- pre-pass: convert one [K x 128] f32 weight to transposed padded bf16 image ----
__global__ __launch_bounds__(kThreads)
void convert_weight_kernel(const float* __restrict__ src, __bf16* __restrict__ dst,
                           int K, int stride) {
  const int i = blockIdx.x * kThreads + threadIdx.x;
  if (i < K * kUnits) {
    const int n = i & (kUnits - 1);
    const int k = i >> 7;
    dst[n * stride + k] = (__bf16)src[i];               // src[k*128 + n] == src[i]
  }
}

__global__ __launch_bounds__(kThreads, 1)
void gru_vae_fused_kernel(const float* __restrict__ x,
                          const float* __restrict__ hidden,
                          const __bf16* __restrict__ wts,
                          const float* __restrict__ bu1, const float* __restrict__ bu2,
                          const float* __restrict__ br1, const float* __restrict__ br2,
                          const float* __restrict__ bn1, const float* __restrict__ bn2,
                          float* __restrict__ out) {
  extern __shared__ __bf16 smem[];
  __bf16* yBuf  = smem;                               // [128][kYS]  y = [hidden | x]
  __bf16* wBuf  = yBuf  + kRowsPerBlock * kYS;        // [128][kWS]  layer-1 weight (transposed)
  __bf16* w2Buf = wBuf  + kUnits * kWS;               // [128][kW2S] layer-2 weight (transposed)
  __bf16* h1Buf = w2Buf + kUnits * kW2S;              // [128][kHS]  tanh(layer-1) activations

  const int tid     = threadIdx.x;
  const int lane    = tid & 31;
  const int rowBase = (tid >> 5) << 4;
  const size_t blockRow = (size_t)blockIdx.x * kRowsPerBlock;

  // ---- kick off gate-u weight DMAs; convert activations to bf16 meanwhile ----
  stage_weight(wBuf,  wts + oWu1, kW1Bytes, tid);
  stage_weight(w2Buf, wts + oWu2, kW2Bytes, tid);

  for (int i = tid; i < kRowsPerBlock * kLatent / 4; i += kThreads) {
    const int idx = i << 2;
    const int row = idx >> 7, col = idx & (kLatent - 1);
    v4f h = *(const v4f*)(hidden + (blockRow + row) * kLatent + col);
    v4bf hb; hb[0] = (__bf16)h[0]; hb[1] = (__bf16)h[1]; hb[2] = (__bf16)h[2]; hb[3] = (__bf16)h[3];
    *(v4bf*)(yBuf + row * kYS + col) = hb;
  }
  for (int i = tid; i < kRowsPerBlock * kInput / 4; i += kThreads) {
    const int idx = i << 2;
    const int row = idx >> 8, col = idx & (kInput - 1);
    v4f v = *(const v4f*)(x + (blockRow + row) * kInput + col);
    v4bf vb; vb[0] = (__bf16)v[0]; vb[1] = (__bf16)v[1]; vb[2] = (__bf16)v[2]; vb[3] = (__bf16)v[3];
    *(v4bf*)(yBuf + row * kYS + kLatent + col) = vb;
  }

  v8f acc[8];
  v8f uFrag[8];   // update gate persists in VGPRs until the final blend

  auto store_h1_tanh = [&](v8f* a) {
#pragma unroll
    for (int nt = 0; nt < 8; ++nt)
#pragma unroll
      for (int j = 0; j < 8; ++j) {
        const int r = rowBase + j + ((lane >> 4) << 3);
        h1Buf[r * kHS + nt * 16 + (lane & 15)] = (__bf16)fast_tanhf_(a[nt][j]);
      }
  };

  weight_fence(tid);
  __syncthreads();                                        // y + Wu1 + Wu2 ready

  // =================== update gate: u = sigmoid(MLP_u(y)) ===================
  gemm_16xN<12>(yBuf, kYS, wBuf, kWS, bu1, rowBase, lane, acc);
  store_h1_tanh(acc);                                     // wave-local rows
  __syncthreads();                                        // wBuf free
  stage_weight(wBuf, wts + oWr1, kW1Bytes, tid);          // DMA Wr1 under gemm2
  gemm_16xN<4>(h1Buf, kHS, w2Buf, kW2S, bu2, rowBase, lane, uFrag);
#pragma unroll
  for (int nt = 0; nt < 8; ++nt)
#pragma unroll
    for (int j = 0; j < 8; ++j) uFrag[nt][j] = sigmoidf_(uFrag[nt][j]);
  __syncthreads();                                        // w2Buf free
  stage_weight(w2Buf, wts + oWr2, kW2Bytes, tid);
  weight_fence(tid);
  __syncthreads();                                        // Wr1 + Wr2 ready

  // =================== reset gate: r = sigmoid(MLP_r(y)); y.hidden *= r =====
  gemm_16xN<12>(yBuf, kYS, wBuf, kWS, br1, rowBase, lane, acc);
  store_h1_tanh(acc);
  __syncthreads();                                        // wBuf free
  stage_weight(wBuf, wts + oWn1, kW1Bytes, tid);          // DMA Wn1 under gemm2
  gemm_16xN<4>(h1Buf, kHS, w2Buf, kW2S, br2, rowBase, lane, acc);
#pragma unroll
  for (int nt = 0; nt < 8; ++nt)       // overwrite hidden columns of yBuf with h*r (wave-local)
#pragma unroll
    for (int j = 0; j < 8; ++j) {
      const int r = rowBase + j + ((lane >> 4) << 3);
      const int c = nt * 16 + (lane & 15);                // c < 128 == kLatent
      __bf16* p = yBuf + r * kYS + c;
      *p = (__bf16)((float)*p * sigmoidf_(acc[nt][j]));
    }
  __syncthreads();                                        // w2Buf free
  stage_weight(w2Buf, wts + oWn2, kW2Bytes, tid);
  weight_fence(tid);
  __syncthreads();                                        // Wn1 + Wn2 ready

  // =================== candidate: n = MLP_n([h*r | x]) ======================
  gemm_16xN<12>(yBuf, kYS, wBuf, kWS, bn1, rowBase, lane, acc);
  store_h1_tanh(acc);                                     // wave-local; in-order DS per wave
  gemm_16xN<4>(h1Buf, kHS, w2Buf, kW2S, bn2, rowBase, lane, acc);

  // =================== blend + store: out = (1-u)*n + u*h (f32) =============
#pragma unroll
  for (int nt = 0; nt < 8; ++nt)
#pragma unroll
    for (int j = 0; j < 8; ++j) {
      const int r = rowBase + j + ((lane >> 4) << 3);
      const int c = nt * 16 + (lane & 15);
      const size_t g = (blockRow + r) * kLatent + c;
      const float u = uFrag[nt][j];
      out[g] = (1.0f - u) * acc[nt][j] + u * hidden[g];
    }
}

extern "C" void kernel_launch(void* const* d_in, const int* in_sizes, int n_in,
                              void* d_out, int out_size, void* d_ws, size_t ws_size,
                              hipStream_t stream) {
  const float* x      = (const float*)d_in[0];
  const float* hidden = (const float*)d_in[1];
  const float* Wu1 = (const float*)d_in[2];  const float* bu1 = (const float*)d_in[3];
  const float* Wu2 = (const float*)d_in[4];  const float* bu2 = (const float*)d_in[5];
  const float* Wr1 = (const float*)d_in[6];  const float* br1 = (const float*)d_in[7];
  const float* Wr2 = (const float*)d_in[8];  const float* br2 = (const float*)d_in[9];
  const float* Wn1 = (const float*)d_in[10]; const float* bn1 = (const float*)d_in[11];
  const float* Wn2 = (const float*)d_in[12]; const float* bn2 = (const float*)d_in[13];
  float* out   = (float*)d_out;
  __bf16* wts  = (__bf16*)d_ws;

  // ---- pre-pass: bf16-convert + transpose all six weight matrices into d_ws ----
  const int w1Blocks = kLI * kUnits / kThreads;     // 192
  const int w2Blocks = kUnits * kUnits / kThreads;  // 64
  convert_weight_kernel<<<w1Blocks, kThreads, 0, stream>>>(Wu1, wts + oWu1, kLI,    kWS);
  convert_weight_kernel<<<w1Blocks, kThreads, 0, stream>>>(Wr1, wts + oWr1, kLI,    kWS);
  convert_weight_kernel<<<w1Blocks, kThreads, 0, stream>>>(Wn1, wts + oWn1, kLI,    kWS);
  convert_weight_kernel<<<w2Blocks, kThreads, 0, stream>>>(Wu2, wts + oWu2, kUnits, kW2S);
  convert_weight_kernel<<<w2Blocks, kThreads, 0, stream>>>(Wr2, wts + oWr2, kUnits, kW2S);
  convert_weight_kernel<<<w2Blocks, kThreads, 0, stream>>>(Wn2, wts + oWn2, kUnits, kW2S);

  const int Brows = in_sizes[0] / kInput;            // 262144
  dim3 grid(Brows / kRowsPerBlock);                  // 2048 workgroups
  gru_vae_fused_kernel<<<grid, kThreads, kShmemBytes, stream>>>(
      x, hidden, wts, bu1, bu2, br1, br2, bn1, bn2, out);
}
